// DispatchEinsumCombine_62878321214338
// MI455X (gfx1250) — matrix operand order repaired and checked by reading, not compiled, not run on hardware
//
#include <hip/hip_runtime.h>
#include <math.h>

// Problem constants (from reference setup_inputs):
#define T_TOK 4096   // B*S tokens
#define H_DIM 1024   // hidden
#define E_NUM 16     // experts
#define I_DIM 1024   // inter
#define N2    2048   // 2*inter
#define KTOP  4

typedef __bf16 bf16_t;
typedef __attribute__((ext_vector_type(16))) __bf16 v16bf;
typedef __attribute__((ext_vector_type(8)))  float  v8f;

// Workspace layout (bytes). Total ~98 MB.
#define OFF_COUNTS 0u
#define OFF_ELIST  1024u
#define OFF_EW     (OFF_ELIST + (unsigned)(E_NUM * T_TOK * 4))
#define OFF_WGU    ((size_t)2 << 20)                                  // 2 MB
#define OFF_WDN    (OFF_WGU + (size_t)E_NUM * H_DIM * N2 * 2)         // +64 MB

__device__ __forceinline__ unsigned short f2bf(float f) {
  unsigned u = __float_as_uint(f);
  u += 0x7fffu + ((u >> 16) & 1u);   // round-to-nearest-even
  return (unsigned short)(u >> 16);
}

// XOR-swizzled LDS index (ushort units): 16B chunks rotated by row to kill
// bank conflicts on 16-lane A-fragment loads. Rows stay 1024 ushorts apart.
__device__ __forceinline__ int lds_idx(int row, int col) {
  return row * 1024 + ((((col >> 3) ^ (row & 7)) << 3) | (col & 7));
}

// ---------------------------------------------------------------------------
// Router: logits -> softmax -> top-4; append (token, weight) to expert lists.
// ---------------------------------------------------------------------------
__global__ __launch_bounds__(256) void router_kernel(
    const float* __restrict__ hs, const float* __restrict__ rw,
    const float* __restrict__ rb, int* __restrict__ counts,
    int* __restrict__ elist, float* __restrict__ ew) {
  int tok = blockIdx.x;
  int tid = threadIdx.x;
  int e = tid >> 4, i = tid & 15;
  const float* x = hs + (size_t)tok * H_DIM;
  const float* w = rw + (size_t)e * H_DIM;
  float p = 0.f;
  for (int h = i; h < H_DIM; h += 16) p += x[h] * w[h];
  __shared__ float red[256];
  __shared__ float logit[E_NUM];
  red[tid] = p;
  __syncthreads();
  for (int s = 8; s; s >>= 1) {
    if (i < s) red[tid] += red[tid + s];
    __syncthreads();
  }
  if (i == 0) logit[e] = red[tid] + rb[e];
  __syncthreads();
  if (tid == 0) {
    float mx = logit[0];
    for (int q = 1; q < E_NUM; ++q) mx = fmaxf(mx, logit[q]);
    float pr[E_NUM];
    float sum = 0.f;
    for (int q = 0; q < E_NUM; ++q) { pr[q] = expf(logit[q] - mx); sum += pr[q]; }
    float inv = 1.f / sum;
    unsigned used = 0;
    for (int k = 0; k < KTOP; ++k) {
      int best = 0; float bv = -1.f;
      for (int q = 0; q < E_NUM; ++q)
        if (!((used >> q) & 1u) && pr[q] > bv) { bv = pr[q]; best = q; }
      used |= 1u << best;
      int pos = atomicAdd(&counts[best], 1);
      elist[best * T_TOK + pos] = tok;
      ew[best * T_TOK + pos] = bv * inv;
    }
  }
}

// ---------------------------------------------------------------------------
// Weight pre-swizzle: f32 [E][K][N] -> bf16 WMMA B fragments.
// frag = [e][k/32][n/16]; lane = (n&15) + 16*((k>>3)&1) owns 16 ushorts, elem = (k&7) + 8*((k>>4)&1).
// One thread per (e, k, n-tile): 4x float4 loads -> 16 scalar stores.
// ---------------------------------------------------------------------------
__global__ void swizzle_kernel(const float* __restrict__ src,
                               unsigned short* __restrict__ dst,
                               int Kd, int Nd) {
  int ntiles = Nd >> 4;
  int ksteps = Kd >> 5;
  size_t total = (size_t)E_NUM * Kd * ntiles;
  for (size_t idx = blockIdx.x * (size_t)blockDim.x + threadIdx.x; idx < total;
       idx += (size_t)gridDim.x * blockDim.x) {
    int nt = (int)(idx % ntiles);
    size_t ek = idx / ntiles;
    int k = (int)(ek % Kd);
    int e = (int)(ek / Kd);
    const float4* s = (const float4*)(src + ((size_t)e * Kd + k) * Nd + nt * 16);
    float4 f0 = s[0], f1 = s[1], f2 = s[2], f3 = s[3];
    float v[16] = {f0.x, f0.y, f0.z, f0.w, f1.x, f1.y, f1.z, f1.w,
                   f2.x, f2.y, f2.z, f2.w, f3.x, f3.y, f3.z, f3.w};
    int kk = k & 31;
    int so = (kk & 7) + 8 * (kk >> 4);
    int hh = (kk >> 3) & 1;
    unsigned short* d =
        dst + ((size_t)(e * ksteps + (k >> 5)) * ntiles + nt) * 512;
    for (int j = 0; j < 16; ++j) d[(j + 16 * hh) * 16 + so] = f2bf(v[j]);
  }
}

// ---------------------------------------------------------------------------
// Fragment loaders
// ---------------------------------------------------------------------------
__device__ __forceinline__ v16bf load_b_frag(const unsigned short* p) {
  union { uint4 q[2]; v16bf v; } u;
  u.q[0] = *(const uint4*)(p);
  u.q[1] = *(const uint4*)(p + 8);
  return u.v;
}

// A 16x32 bf16 fragment from swizzled LDS:
// lane<16 holds K {0..7,16..23}, lane>=16 holds K {8..15,24..31} of the chunk.
__device__ __forceinline__ v16bf load_a_frag(const unsigned short* buf, int row, int klo) {
  union { uint4 q[2]; v16bf v; } u;
  u.q[0] = *(const uint4*)(buf + lds_idx(row, klo));
  u.q[1] = *(const uint4*)(buf + lds_idx(row, klo + 16));
  return u.v;
}

__device__ __forceinline__ v8f zero8() {
  v8f z = {0.f, 0.f, 0.f, 0.f, 0.f, 0.f, 0.f, 0.f};
  return z;
}

// ---------------------------------------------------------------------------
// Fused expert MLP: 16-token tile x 1 expert per workgroup, 8 waves.
//   phase1: GU = hsT(16x1024) @ Wgu(1024x2048), GLU activation -> actT (LDS)
//   phase2: OUT = actT(16x1024) @ Wdn(1024x1024), *topk_w, fp-atomic into out
// ---------------------------------------------------------------------------
__global__ __launch_bounds__(256) void moe_kernel(
    const float* __restrict__ hs,
    const unsigned short* __restrict__ wgu,
    const unsigned short* __restrict__ wdn,
    const float* __restrict__ gub,
    const float* __restrict__ db,
    const int* __restrict__ counts,
    const int* __restrict__ elist,
    const float* __restrict__ ew,
    float* __restrict__ out) {
  int e = blockIdx.x >> 8;
  int tile = blockIdx.x & 255;
  int cnt = counts[e];
  int base = tile * 16;
  if (base >= cnt) return;   // block-uniform: EXEC stays all-ones below

  __shared__ unsigned short hsT[16 * 1024];   // 32 KB bf16 (swizzled)
  __shared__ unsigned short actT[16 * 1024];  // 32 KB bf16 (swizzled)

  int tid = threadIdx.x;

  // Stage hidden tile: 8 floats -> one 16B swizzled LDS chunk per iteration.
  for (int i = tid; i < 16 * (H_DIM / 8); i += 256) {
    int r = i >> 7, cc = i & 127;
    int idx = base + r;
    int tok = (idx < cnt) ? elist[e * T_TOK + idx] : 0;
    const float4* s = (const float4*)(hs + (size_t)tok * H_DIM + cc * 8);
    float4 f0 = s[0], f1 = s[1];
    uint4 p;
    p.x = f2bf(f0.x) | ((unsigned)f2bf(f0.y) << 16);
    p.y = f2bf(f0.z) | ((unsigned)f2bf(f0.w) << 16);
    p.z = f2bf(f1.x) | ((unsigned)f2bf(f1.y) << 16);
    p.w = f2bf(f1.z) | ((unsigned)f2bf(f1.w) << 16);
    *(uint4*)(hsT + lds_idx(r, cc * 8)) = p;
  }
  __syncthreads();

  int wv = tid >> 5, lane = tid & 31;
  int m0 = lane & 15, khalf = lane >> 4;

  // ---- phase 1: gate_up GEMM ----
  v8f cg[8], cu[8];
#pragma unroll
  for (int t = 0; t < 8; ++t) { cg[t] = zero8(); cu[t] = zero8(); }

  for (int ks = 0; ks < 32; ++ks) {
    v16bf a = load_a_frag(hsT, m0, ks * 32 + khalf * 8);
    size_t fb = (size_t)(e * 32 + ks) * 128;  // 128 n-tiles across N2
    const unsigned short* pg = wgu + (fb + wv * 8) * 512 + lane * 16;
    const unsigned short* pu = wgu + (fb + 64 + wv * 8) * 512 + lane * 16;
    v16bf bg[8];
#pragma unroll
    for (int nt = 0; nt < 8; ++nt) bg[nt] = load_b_frag(pg + nt * 512);
#pragma unroll
    for (int nt = 0; nt < 8; ++nt)
      cg[nt] = __builtin_amdgcn_wmma_f32_16x16x32_bf16(
          false, a, false, bg[nt], (short)0, cg[nt], false, false);
    v16bf bu[8];
#pragma unroll
    for (int nt = 0; nt < 8; ++nt) bu[nt] = load_b_frag(pu + nt * 512);
#pragma unroll
    for (int nt = 0; nt < 8; ++nt)
      cu[nt] = __builtin_amdgcn_wmma_f32_16x16x32_bf16(
          false, a, false, bu[nt], (short)0, cu[nt], false, false);
  }

  // ---- GLU activation -> actT ----
#pragma unroll
  for (int nt = 0; nt < 8; ++nt) {
    int cG = wv * 128 + nt * 16 + m0;   // 0..1023 (gate col == up col - 1024)
    float bg = gub[e * N2 + cG];
    float bu = gub[e * N2 + I_DIM + cG];
#pragma unroll
    for (int j = 0; j < 8; ++j) {
      int m = j + 8 * khalf;
      float g = cg[nt][j] + bg;
      float u = cu[nt][j] + bu;
      g = fminf(g, 7.f);
      u = fminf(fmaxf(u, -7.f), 7.f);
      float glu = g / (1.f + __expf(-1.702f * g));
      actT[lds_idx(m, cG)] = f2bf((u + 1.f) * glu);
    }
  }
  __syncthreads();

  // ---- phase 2: down GEMM ----
  v8f co[8];
#pragma unroll
  for (int t = 0; t < 8; ++t) co[t] = zero8();
#pragma unroll 2
  for (int ks = 0; ks < 32; ++ks) {
    v16bf a = load_a_frag(actT, m0, ks * 32 + khalf * 8);
    size_t fb = (size_t)(e * 32 + ks) * 64;  // 64 n-tiles across H
    const unsigned short* pd = wdn + (fb + wv * 8) * 512 + lane * 16;
    v16bf b[8];
#pragma unroll
    for (int nt = 0; nt < 8; ++nt) b[nt] = load_b_frag(pd + nt * 512);
#pragma unroll
    for (int nt = 0; nt < 8; ++nt)
      co[nt] = __builtin_amdgcn_wmma_f32_16x16x32_bf16(
          false, a, false, b[nt], (short)0, co[nt], false, false);
  }

  // ---- combine: (acc + down_bias) * topk_w, scatter-add ----
  int tokR[8];
  float wgtR[8];
#pragma unroll
  for (int j = 0; j < 8; ++j) {
    int idx = base + j + 8 * khalf;
    if (idx < cnt) { tokR[j] = elist[e * T_TOK + idx]; wgtR[j] = ew[e * T_TOK + idx]; }
    else           { tokR[j] = 0; wgtR[j] = 0.f; }
  }
#pragma unroll
  for (int nt = 0; nt < 8; ++nt) {
    int col = wv * 128 + nt * 16 + m0;
    float bias = db[e * H_DIM + col];
#pragma unroll
    for (int j = 0; j < 8; ++j) {
      float wgt = wgtR[j];
      if (wgt != 0.f) {
        // Device-scope relaxed FP atomic -> native global_atomic_add_f32.
        __hip_atomic_fetch_add(out + (size_t)tokR[j] * H_DIM + col,
                               (co[nt][j] + bias) * wgt,
                               __ATOMIC_RELAXED, __HIP_MEMORY_SCOPE_AGENT);
      }
    }
  }
}

// ---------------------------------------------------------------------------
extern "C" void kernel_launch(void* const* d_in, const int* in_sizes, int n_in,
                              void* d_out, int out_size, void* d_ws, size_t ws_size,
                              hipStream_t stream) {
  const float* hs  = (const float*)d_in[0];
  const float* rw  = (const float*)d_in[1];
  const float* rb  = (const float*)d_in[2];
  const float* gup = (const float*)d_in[3];
  const float* gub = (const float*)d_in[4];
  const float* dnp = (const float*)d_in[5];
  const float* dnb = (const float*)d_in[6];
  float* out = (float*)d_out;

  char* ws = (char*)d_ws;
  int* counts         = (int*)(ws + OFF_COUNTS);
  int* elist          = (int*)(ws + OFF_ELIST);
  float* ew           = (float*)(ws + OFF_EW);
  unsigned short* wgu = (unsigned short*)(ws + OFF_WGU);
  unsigned short* wdn = (unsigned short*)(ws + OFF_WDN);

  (void)hipMemsetAsync(d_out, 0, (size_t)out_size * sizeof(float), stream);
  (void)hipMemsetAsync(counts, 0, 1024, stream);

  router_kernel<<<T_TOK, 256, 0, stream>>>(hs, rw, rb, counts, elist, ew);
  swizzle_kernel<<<4096, 256, 0, stream>>>(gup, wgu, H_DIM, N2);
  swizzle_kernel<<<4096, 256, 0, stream>>>(dnp, wdn, I_DIM, H_DIM);
  moe_kernel<<<E_NUM * 256, 256, 0, stream>>>(hs, wgu, wdn, gub, dnb,
                                              counts, elist, ew, out);
}
